// Attention_56435870270088
// MI455X (gfx1250) — compile-verified, matrix operand order checked by reading, not run on hardware
//
#include <hip/hip_runtime.h>
#include <cstdint>

// ---------------------------------------------------------------------------
// MI455X (gfx1250) attention layer: bf16 WMMA everywhere.
//   q = rope(x@wq), k = rope(x@wk), v = x@wv   (bf16 in workspace)
//   flash attention (causal, online softmax)    (bf16 in workspace)
//   out = attn @ wo                             (fp32 to d_out, A tile via TDM)
// ---------------------------------------------------------------------------

typedef __attribute__((ext_vector_type(16))) __bf16        v16bf;
typedef __attribute__((ext_vector_type(8)))  float         v8f;
typedef __attribute__((ext_vector_type(8)))  unsigned int  v8u;
typedef __attribute__((ext_vector_type(4)))  unsigned int  u32x4;
typedef __attribute__((ext_vector_type(8)))  int           i32x8;
typedef __attribute__((ext_vector_type(4)))  int           i32x4;

#if defined(__has_builtin)
#  if __has_builtin(__builtin_amdgcn_tensor_load_to_lds)
#    define HAVE_TDM 1
#  else
#    define HAVE_TDM 0
#  endif
#  if __has_builtin(__builtin_amdgcn_s_wait_tensorcnt)
#    define HAVE_WTC 1
#  else
#    define HAVE_WTC 0
#  endif
#else
#  define HAVE_TDM 0
#  define HAVE_WTC 0
#endif

__device__ __forceinline__ void wait_tensorcnt0() {
#if HAVE_WTC
    __builtin_amdgcn_s_wait_tensorcnt(0);
#else
    asm volatile("s_wait_tensorcnt 0x0" ::: "memory");
#endif
}

// hardware bf16 converts (v_cvt_*): much cheaper than manual RNE bit-twiddling
__device__ __forceinline__ uint16_t f2bf(float f) {
    __bf16 b = (__bf16)f;
    return __builtin_bit_cast(uint16_t, b);
}
__device__ __forceinline__ float bf2f(uint16_t h) {
    return (float)__builtin_bit_cast(__bf16, h);
}

// D = A(16x32 bf16) * B(32x16 bf16) + C(16x16 f32)
__device__ __forceinline__ v8f wmma_bf16(v16bf a, v16bf b, v8f c) {
    return __builtin_amdgcn_wmma_f32_16x16x32_bf16(
        /*neg_a=*/false, a, /*neg_b=*/false, b,
        /*c_mod=*/(short)0, c, /*reuse_a=*/false, /*reuse_b=*/false);
}

// Load one 16x32 bf16 fragment in the CDNA5 WMMA A/B register layout.
// Per ISA 7.12.2: lanes 0-15 hold row r, K = 0..7 & 16..23; lanes 16-31 hold
// row r, K = 8..15 & 24..31 -> two contiguous 16B loads per lane.
__device__ __forceinline__ v16bf load_frag(const uint16_t* base, int stride_halfs) {
    const int lane = (int)(threadIdx.x & 31u);
    const int r  = lane & 15;
    const int hi = lane >> 4;
    const uint16_t* p = base + (size_t)r * stride_halfs + hi * 8;
    uint4 a = *reinterpret_cast<const uint4*>(p);
    uint4 b = *reinterpret_cast<const uint4*>(p + 16);
    v8u u;
    u[0] = a.x; u[1] = a.y; u[2] = a.z; u[3] = a.w;
    u[4] = b.x; u[5] = b.y; u[6] = b.z; u[7] = b.w;
    return __builtin_bit_cast(v16bf, u);
}

// ---------------------------------------------------------------------------
// Tiled WMMA GEMM: C(MxN) = A(MxK) * B(KxN).  A fp32-or-bf16, B fp32 (cvt on
// load), C bf16-or-fp32.  Block tile 128x128, BK=32, 256 thr = 8 waves (2x4),
// each wave computes 64x32 via 4x2 accumulators.  When A is bf16, the A tile
// is staged into LDS by the Tensor Data Mover (pad fields produce the 80B
// padded LDS rows directly).
// ---------------------------------------------------------------------------
constexpr int BM = 128, BN = 128, BK = 32, LDT = 40;  // LDT: pad to 80B rows

template <bool A_BF16, bool OUT_BF16>
__global__ __launch_bounds__(256) void gemm_wmma_kernel(
    const void* __restrict__ Av, const float* __restrict__ B,
    void* __restrict__ Cv, int M, int N, int K, int lda, int ldb, int ldc)
{
    __shared__ uint16_t As[BM * LDT];
    __shared__ uint16_t Bs[BN * LDT];   // stored transposed: [n][k]

    const int tid  = (int)threadIdx.x;
    const int lane = tid & 31, wave = tid >> 5;
    const int wm = wave & 1, wn = wave >> 1;        // 2 x 4 wave grid
    const int row0 = (int)blockIdx.y * BM;
    const int col0 = (int)blockIdx.x * BN;
    const int hi = lane >> 4, ln = lane & 15;

    v8f acc[4][2];
    for (int i = 0; i < 4; i++)
        for (int j = 0; j < 2; j++)
            for (int e = 0; e < 8; e++) acc[i][j][e] = 0.f;

    const float*    Af = (const float*)Av;
    const uint16_t* Ah = (const uint16_t*)Av;

    for (int k0 = 0; k0 < K; k0 += BK) {
        // ---- stage A tile (row-major bf16, LDS rows padded to LDT halfs) ----
        if (A_BF16) {
#if HAVE_TDM
            if (tid < 32) {   // one wave issues the TDM transfer (EXEC ignored)
                const uint64_t ga = (uint64_t)(uintptr_t)Ah +
                                    ((uint64_t)row0 * (uint64_t)lda + (uint64_t)k0) * 2u;
                const uint32_t ldsa = (uint32_t)(uintptr_t)(void*)As; // LDS byte offset
                u32x4 g0;
                g0[0] = 1u;                                   // count=1 (valid user D#)
                g0[1] = ldsa;                                 // lds_addr
                g0[2] = (uint32_t)(ga & 0xFFFFFFFFu);         // global_addr[31:0]
                g0[3] = (uint32_t)((ga >> 32) & 0x1FFFFFFu)   // global_addr[56:32]
                      | (2u << 30);                           // type=2 ("image")
                const uint32_t td0 = (uint32_t)lda;           // tensor dim0 (elements)
                const uint32_t td1 = (uint32_t)M;             // tensor dim1 (rows)
                i32x8 g1;
                g1[0] = (int)((1u << 16)      // data_size = 2 bytes
                            | (1u << 20)      // pad_enable
                            | (3u << 22)      // pad_interval: every 16 DWORDs (=32 halfs)
                            | (3u << 25));    // pad_amount: 4 DWORDs (=8 halfs) -> LDT=40
                g1[1] = (int)((td0 & 0xFFFFu) << 16);               // tensor_dim0[15:0]
                g1[2] = (int)((td0 >> 16) | ((td1 & 0xFFFFu) << 16)); // dim0 hi | dim1 lo
                g1[3] = (int)((td1 >> 16) | ((uint32_t)BK << 16));  // dim1 hi | tile_dim0=32
                g1[4] = (int)(uint32_t)BM;                          // tile_dim1=128, tile_dim2=0
                g1[5] = (int)(uint32_t)lda;                         // tensor_dim0_stride[31:0]
                g1[6] = 0;                                          // stride hi | dim1_stride lo
                g1[7] = 0;
                i32x4 g2; i32x4 g3;
                for (int i = 0; i < 4; i++) { g2[i] = 0; g3[i] = 0; }
#if __clang_major__ >= 23
                i32x8 g4;
                for (int i = 0; i < 8; i++) g4[i] = 0;
                __builtin_amdgcn_tensor_load_to_lds(g0, g1, g2, g3, g4, 0);
#else
                __builtin_amdgcn_tensor_load_to_lds(g0, g1, g2, g3, 0);
#endif
                wait_tensorcnt0();
            }
#else
            for (int t = 0; t < 2; t++) {                 // 512 x 16B fallback copy
                int idx = tid + t * 256;
                int r = idx >> 2, c8 = idx & 3;
                uint4 h = *reinterpret_cast<const uint4*>(
                    Ah + (size_t)(row0 + r) * lda + k0 + c8 * 8);
                *reinterpret_cast<uint4*>(&As[r * LDT + c8 * 8]) = h;
            }
#endif
        } else {
            for (int t = 0; t < 4; t++) {                 // 1024 x float4
                int idx = tid + t * 256;
                int r = idx >> 3, c4 = idx & 7;
                float4 f = *reinterpret_cast<const float4*>(
                    Af + (size_t)(row0 + r) * lda + k0 + c4 * 4);
                uint16_t* d = &As[r * LDT + c4 * 4];
                d[0] = f2bf(f.x); d[1] = f2bf(f.y);
                d[2] = f2bf(f.z); d[3] = f2bf(f.w);
            }
        }
        // ---- stage B tile transposed ([n][k] bf16) ----
        for (int t = 0; t < 4; t++) {                     // 1024 x float4
            int idx = tid + t * 256;
            int r = idx >> 5, c4 = idx & 31;
            float4 f = *reinterpret_cast<const float4*>(
                B + (size_t)(k0 + r) * ldb + col0 + c4 * 4);
            Bs[(c4 * 4 + 0) * LDT + r] = f2bf(f.x);
            Bs[(c4 * 4 + 1) * LDT + r] = f2bf(f.y);
            Bs[(c4 * 4 + 2) * LDT + r] = f2bf(f.z);
            Bs[(c4 * 4 + 3) * LDT + r] = f2bf(f.w);
        }
        // prefetch next K-tile while WMMAs run (global_prefetch_b8)
        if (k0 + BK < K) {
            __builtin_prefetch(B + (size_t)(k0 + BK + (tid >> 5)) * ldb + col0 + (tid & 31) * 4, 0, 3);
            if (!A_BF16)
                __builtin_prefetch(Af + (size_t)(row0 + (tid >> 1)) * lda + k0 + BK, 0, 3);
        }
        __syncthreads();

        v16bf af[4], bfr[2];
        for (int im = 0; im < 4; im++)
            af[im] = load_frag(&As[(wm * 64 + im * 16) * LDT], LDT);
        for (int jn = 0; jn < 2; jn++)
            bfr[jn] = load_frag(&Bs[(wn * 32 + jn * 16) * LDT], LDT);
        for (int im = 0; im < 4; im++)
            for (int jn = 0; jn < 2; jn++)
                acc[im][jn] = wmma_bf16(af[im], bfr[jn], acc[im][jn]);
        __syncthreads();
    }

    // ---- epilogue: C layout = VGPR e holds row e(+8 for hi half), col = ln
    for (int im = 0; im < 4; im++)
        for (int jn = 0; jn < 2; jn++) {
            int col = col0 + wn * 32 + jn * 16 + ln;
            for (int e = 0; e < 8; e++) {
                int row = row0 + wm * 64 + im * 16 + e + 8 * hi;
                float val = acc[im][jn][e];
                if (OUT_BF16)
                    ((uint16_t*)Cv)[(size_t)row * ldc + col] = f2bf(val);
                else
                    ((float*)Cv)[(size_t)row * ldc + col] = val;
            }
        }
}

// ---------------------------------------------------------------------------
// RoPE on bf16 q (2048x4096, 32 heads) and k (2048x2048, 16 heads), in place.
// ---------------------------------------------------------------------------
__global__ __launch_bounds__(256) void rope_kernel(
    uint16_t* __restrict__ q, uint16_t* __restrict__ k,
    const float* __restrict__ fc, const int* __restrict__ start_pos)
{
    const int QP = 2048 * 2048;     // q rotation pairs
    const int KP = 2048 * 1024;     // k rotation pairs
    int idx = (int)(blockIdx.x * blockDim.x + threadIdx.x);
    if (idx >= QP + KP) return;

    uint16_t* t;
    int pos, rem;
    if (idx < QP) { pos = idx >> 11; rem = idx & 2047; t = q + (size_t)pos * 4096 + rem * 2; }
    else { int i2 = idx - QP; pos = i2 >> 10; rem = i2 & 1023; t = k + (size_t)pos * 2048 + rem * 2; }
    int j = rem & 63;                              // pair index within head_dim/2
    int p = *start_pos + pos;
    float c = fc[(size_t)(p * 64 + j) * 2 + 0];
    float s = fc[(size_t)(p * 64 + j) * 2 + 1];
    float t0 = bf2f(t[0]), t1 = bf2f(t[1]);
    t[0] = f2bf(t0 * c - t1 * s);
    t[1] = f2bf(t0 * s + t1 * c);
}

// ---------------------------------------------------------------------------
// Flash attention (causal).  Grid (32 qblocks, 32 heads), 128 thr = 4 waves.
// Each wave owns 16 query rows; loops 32-key tiles with online softmax.
// ---------------------------------------------------------------------------
constexpr int HD  = 128;
constexpr int VST = 40;   // V-transpose LDS row pad (80B: aligned + no conflicts)

__global__ __launch_bounds__(128) void flash_attn_kernel(
    const uint16_t* __restrict__ q, const uint16_t* __restrict__ k,
    const uint16_t* __restrict__ v, uint16_t* __restrict__ o)
{
    __shared__ uint16_t Pst[4][16 * 32];     // per-wave P staging (A layout src)
    __shared__ uint16_t Vt[4][HD * VST];     // per-wave V tile, transposed [feat][key]

    const int tid = (int)threadIdx.x, lane = tid & 31, wave = tid >> 5;
    const int h = (int)blockIdx.y, kvh = h >> 1;
    const int qbase = (int)blockIdx.x * 64 + wave * 16;
    const int hi = lane >> 4, ln = lane & 15;
    const float scale = 0.0883883476483184405f;   // 1/sqrt(128)

    // Q fragments: 16 rows x 128 head-dims = 4 K-chunks of 32
    v16bf qf[4];
    for (int c = 0; c < 4; c++)
        qf[c] = load_frag(q + (size_t)qbase * 4096 + h * HD + c * 32, 4096);

    float m_run[8], l_run[8];
    v8f O[8];
    for (int e = 0; e < 8; e++) {
        m_run[e] = -1e30f; l_run[e] = 0.f;
        for (int c = 0; c < 8; c++) O[c][e] = 0.f;
    }

    const int nkb = (qbase + 16 + 31) / 32;       // causal tile count for this wave
    for (int kb = 0; kb < nkb; kb++) {
        const int kbase = kb * 32;

        // --- scores: two 16-key halves, 4 WMMAs each over head_dim=128 ---
        float sc[2][8], p[2][8], t[8];
        for (int e = 0; e < 8; e++) t[e] = -1e30f;
        for (int ns = 0; ns < 2; ns++) {
            v16bf kf[4];
            for (int c = 0; c < 4; c++)
                kf[c] = load_frag(k + (size_t)(kbase + ns * 16) * 2048 + kvh * HD + c * 32, 2048);
            v8f S;
            for (int e = 0; e < 8; e++) S[e] = 0.f;
            for (int c = 0; c < 4; c++) S = wmma_bf16(qf[c], kf[c], S);
            const int kidx = kbase + ns * 16 + ln;
            for (int e = 0; e < 8; e++) {
                int qidx = qbase + e + 8 * hi;
                float s = S[e] * scale;
                if (kidx > qidx) s = -1e30f;      // causal mask
                sc[ns][e] = s;
                t[e] = fmaxf(t[e], s);
            }
        }
        // row-wise max across the 16-lane half (wave32: xor masks 1..8)
        for (int m = 1; m < 16; m <<= 1)
            for (int e = 0; e < 8; e++) t[e] = fmaxf(t[e], __shfl_xor(t[e], m, 32));

        float alpha[8], rs[8];
        for (int e = 0; e < 8; e++) {
            float mn = fmaxf(m_run[e], t[e]);
            alpha[e] = __expf(m_run[e] - mn);
            m_run[e] = mn;
            p[0][e] = __expf(sc[0][e] - mn);
            p[1][e] = __expf(sc[1][e] - mn);
            rs[e] = p[0][e] + p[1][e];
        }
        for (int m = 1; m < 16; m <<= 1)
            for (int e = 0; e < 8; e++) rs[e] += __shfl_xor(rs[e], m, 32);
        for (int e = 0; e < 8; e++) l_run[e] = l_run[e] * alpha[e] + rs[e];

        // --- stage P (C layout -> row-major 16x32 bf16) ---
        uint16_t* pp = Pst[wave];
        for (int ns = 0; ns < 2; ns++)
            for (int e = 0; e < 8; e++)
                pp[(e + 8 * hi) * 32 + ns * 16 + ln] = f2bf(p[ns][e]);

        // --- stage V tile transposed: Vt[feat][key], coalesced 16B reads ---
        uint16_t* vt = Vt[wave];
        for (int i = 0; i < 16; i++) {
            int flat = lane + i * 32;             // 512 x 16B over 32x128 tile
            int key = flat >> 4, c8 = flat & 15;
            uint4 hv = *reinterpret_cast<const uint4*>(
                v + (size_t)(kbase + key) * 2048 + kvh * HD + c8 * 8);
            const uint16_t* hp = reinterpret_cast<const uint16_t*>(&hv);
            for (int j = 0; j < 8; j++) vt[(c8 * 8 + j) * VST + key] = hp[j];
        }
        asm volatile("s_wait_dscnt 0x0" ::: "memory");   // wave-private LDS RAW

        // --- O = P*V + alpha*O : 8 WMMAs over the 8 feature chunks ---
        v16bf pf = load_frag(pp, 32);
        for (int c = 0; c < 8; c++) {
            v16bf vf = load_frag(&vt[(c * 16) * VST], VST);
            v8f Cacc;
            for (int e = 0; e < 8; e++) Cacc[e] = O[c][e] * alpha[e];
            O[c] = wmma_bf16(pf, vf, Cacc);
        }
    }

    // --- normalize + write bf16 ---
    for (int c = 0; c < 8; c++)
        for (int e = 0; e < 8; e++) {
            int row = qbase + e + 8 * hi;
            int col = h * HD + c * 16 + ln;
            o[(size_t)row * 4096 + col] = f2bf(O[c][e] / l_run[e]);
        }
}

// ---------------------------------------------------------------------------
// Host-side launcher. Workspace layout (bf16 halves), total 48 MB:
//   q: 2048*4096 | k: 2048*2048 | v: 2048*2048 | attn: 2048*4096
// ---------------------------------------------------------------------------
extern "C" void kernel_launch(void* const* d_in, const int* in_sizes, int n_in,
                              void* d_out, int out_size, void* d_ws, size_t ws_size,
                              hipStream_t stream)
{
    (void)in_sizes; (void)n_in; (void)out_size; (void)ws_size;
    const float* x  = (const float*)d_in[0];
    const float* fc = (const float*)d_in[1];
    const float* wq = (const float*)d_in[2];
    const float* wk = (const float*)d_in[3];
    const float* wv = (const float*)d_in[4];
    const float* wo = (const float*)d_in[5];
    const int*   sp = (const int*)d_in[6];

    uint16_t* qb = (uint16_t*)d_ws;
    uint16_t* kb = qb + (size_t)2048 * 4096;
    uint16_t* vb = kb + (size_t)2048 * 2048;
    uint16_t* ab = vb + (size_t)2048 * 2048;
    float* out = (float*)d_out;

    // Q/K/V projections (fp32 in, bf16 out)
    gemm_wmma_kernel<false, true><<<dim3(32, 16), 256, 0, stream>>>(
        x, wq, qb, 2048, 4096, 4096, 4096, 4096, 4096);
    gemm_wmma_kernel<false, true><<<dim3(16, 16), 256, 0, stream>>>(
        x, wk, kb, 2048, 2048, 4096, 4096, 2048, 2048);
    gemm_wmma_kernel<false, true><<<dim3(16, 16), 256, 0, stream>>>(
        x, wv, vb, 2048, 2048, 4096, 4096, 2048, 2048);

    // RoPE on q and k
    const int total_pairs = 2048 * 2048 + 2048 * 1024;
    rope_kernel<<<(total_pairs + 255) / 256, 256, 0, stream>>>(qb, kb, fc, sp);

    // causal flash attention
    flash_attn_kernel<<<dim3(32, 32), 128, 0, stream>>>(qb, kb, vb, ab);

    // output projection (bf16 in via TDM, fp32 out)
    gemm_wmma_kernel<true, false><<<dim3(32, 16), 256, 0, stream>>>(
        ab, wo, out, 2048, 4096, 4096, 4096, 4096, 4096);
}
